// SGBlock_26371099198354
// MI455X (gfx1250) — matrix operand order repaired and checked by reading, not compiled, not run on hardware
//
#include <hip/hip_runtime.h>
#include <math.h>

// ---------------------------------------------------------------------------
// Shapes: B=256, N=320 (64 "z" tokens on 8x8 grid + 256 "x" tokens on 16x16),
// D=768, HID=8.  Memory-bound: ~504MB traffic @23.3TB/s -> ~22us floor.
// ---------------------------------------------------------------------------

#define DIMD   768
#define HID    8
#define NTOK   320
#define MROWS  (256 * 320)          // 81920

typedef __attribute__((ext_vector_type(16))) __bf16        v16bf;
typedef __attribute__((ext_vector_type(8)))  float         v8f;
typedef __attribute__((ext_vector_type(8)))  unsigned int  v8u;

union Frag { v8u u; v16bf bf; };

// f32 pair -> packed bf16 (round-to-nearest-ish): one v_perm_b32 + two adds.
__device__ __forceinline__ unsigned int pack_bf16(float lo, float hi) {
    unsigned int bl = __float_as_uint(lo) + 0x8000u;
    unsigned int bh = __float_as_uint(hi) + 0x8000u;
    return __builtin_amdgcn_perm(bh, bl, 0x07060302u);  // {bh[31:16], bl[31:16]}
}

// ===========================================================================
// Kernel 1: h = x @ Wd + bd   (81920x768 @ 768x8)  via v_wmma_f32_16x16x32_bf16
// Wd (24KB) staged into LDS with async global->LDS copies (ASYNCcnt path),
// then repacked into WMMA-B layout (24 K-chunks x 8 VGPRs x 32 lanes).
// 4 waves/block, each wave owns a 16-row M tile.
// ===========================================================================
__global__ __launch_bounds__(128) void down_proj_kernel(
    const float* __restrict__ x, const float* __restrict__ Wd,
    const float* __restrict__ bd, float* __restrict__ h)
{
    __shared__ float        ldsWd[DIMD * HID];   // 24 KB raw f32
    __shared__ unsigned int ldsB [24 * 8 * 32];  // 24 KB packed bf16 B frags
    const int tid = threadIdx.x;

    // --- Stage Wd into LDS via async copy (global_load_async_to_lds_b32). ---
    {
        const unsigned ldsBase = (unsigned)(size_t)(void*)ldsWd; // LDS offset
        for (int i = tid; i < DIMD * HID; i += 128) {
            unsigned loff = ldsBase + (unsigned)i * 4u;
            unsigned goff = (unsigned)i * 4u;
            asm volatile("global_load_async_to_lds_b32 %0, %1, %2"
                         :: "v"(loff), "v"(goff), "s"(Wd) : "memory");
        }
        asm volatile("s_wait_asynccnt 0x0" ::: "memory");
    }
    __syncthreads();

    // --- Repack: B(32x16) layout: lane=N (n=l%16), half=l/16, VGPR v holds
    //     K = half*16 + 2v, 2v+1 (cols n>=8 zero-padded). ---
    for (int s = tid; s < 24 * 8 * 32; s += 128) {
        int kc = s >> 8;
        int r  = s & 255;
        int v  = r >> 5;
        int l  = r & 31;
        int n  = l & 15;
        int hf = l >> 4;
        int k  = kc * 32 + hf * 16 + v * 2;
        float lo = (n < HID) ? ldsWd[k * HID + n]       : 0.0f;
        float hi = (n < HID) ? ldsWd[(k + 1) * HID + n] : 0.0f;
        ldsB[s] = pack_bf16(lo, hi);
    }
    __syncthreads();

    const int wave = tid >> 5;
    const int lane = tid & 31;
    const int m    = lane & 15;
    const int half = lane >> 4;
    const int row0 = (blockIdx.x * 4 + wave) * 16;
    const float* xrow = x + (long)(row0 + m) * DIMD;

    v8f acc = {};
    for (int kc = 0; kc < 24; ++kc) {
        if (kc < 23) __builtin_prefetch(xrow + (kc + 1) * 32, 0, 3); // global_prefetch_b8
        Frag a, b;
        #pragma unroll
        for (int v = 0; v < 8; ++v) {
            // A(16x32) layout: K = half*16 + 2v, 2v+1 (matches B repack)
            int kk = kc * 32 + half * 16 + v * 2;
            a.u[v] = pack_bf16(xrow[kk], xrow[kk + 1]);
            b.u[v] = ldsB[kc * 256 + v * 32 + lane];
        }
        acc = __builtin_amdgcn_wmma_f32_16x16x32_bf16(
            false, a.bf, false, b.bf, (short)0, acc, false, false);
    }

    // C/D layout: lanes 0-15 -> N=lane, VGPR i -> M=i; lanes 16-31 -> M=8+i.
    const int n = lane & 15;
    if (n < HID) {
        float bias = bd[n];
        #pragma unroll
        for (int i = 0; i < 8; ++i) {
            int mm = half * 8 + i;
            h[(long)(row0 + mm) * HID + n] = acc[i] + bias;
        }
    }
}

// ===========================================================================
// Kernel 2: fused branch.  One block per image (512 blocks: 256 "x" 16x16,
// 256 "z" 8x8).  LN1 -> full HxH DFT (ortho) -> Hermitian-extended filter ->
// IDFT -> LN2 -> GELU MLP -> +raw.  Only H distinct twiddles (phase mod H).
// ===========================================================================
#define L1G 0
#define L1B 8
#define L2G 16
#define L2B 24
#define W1O 32
#define B1O 160
#define W2O 176
#define B2O 304
#define NPARAM 312

__global__ __launch_bounds__(256) void branch_kernel(
    const float* __restrict__ h,
    const float* __restrict__ cw_x, const float* __restrict__ cw_z,
    const float* __restrict__ ln1_g, const float* __restrict__ ln1_b,
    const float* __restrict__ ln2_g, const float* __restrict__ ln2_b,
    const float* __restrict__ W1, const float* __restrict__ b1,
    const float* __restrict__ W2, const float* __restrict__ b2,
    float* __restrict__ out)
{
    const int  tid    = threadIdx.x;
    const bool isX    = blockIdx.x < 256;
    const int  img    = isX ? blockIdx.x : blockIdx.x - 256;
    const int  H      = isX ? 16 : 8;
    const int  HH     = H * H;
    const int  HB     = H / 2 + 1;
    const int  tokOff = isX ? 64 : 0;
    const float* cw   = isX ? cw_x : cw_z;
    const float  inv  = 1.0f / (float)H;

    __shared__ float Y  [256 * 9];
    __shared__ float Gre[256 * 9];
    __shared__ float Gim[256 * 9];
    __shared__ float tcs[16], tsn[16];
    __shared__ float P[NPARAM];

    for (int i = tid; i < NPARAM; i += 256) {
        float v;
        if      (i < 8)   v = ln1_g[i];
        else if (i < 16)  v = ln1_b[i - 8];
        else if (i < 24)  v = ln2_g[i - 16];
        else if (i < 32)  v = ln2_b[i - 24];
        else if (i < 160) v = W1[i - W1O];
        else if (i < 176) v = b1[i - B1O];
        else if (i < 304) v = W2[i - W2O];
        else              v = b2[i - B2O];
        P[i] = v;
    }
    if (tid < H) {
        float s, c;
        __sincosf(6.283185307179586f * (float)tid * inv, &s, &c);
        tcs[tid] = c; tsn[tid] = s;
    }

    float raw[HID] = {0.f};
    if (tid < HH) {
        const float* src = h + (long)(img * NTOK + tokOff + tid) * HID;
        float mean = 0.f;
        #pragma unroll
        for (int c = 0; c < HID; ++c) { raw[c] = src[c]; mean += raw[c]; }
        mean *= 0.125f;
        float var = 0.f;
        #pragma unroll
        for (int c = 0; c < HID; ++c) { float d = raw[c] - mean; var += d * d; }
        float rstd = rsqrtf(var * 0.125f + 1e-5f);
        #pragma unroll
        for (int c = 0; c < HID; ++c)
            Y[tid * 9 + c] = (raw[c] - mean) * rstd * P[L1G + c] + P[L1B + c];
    }
    __syncthreads();

    // Forward DFT over (ww-axis u, hh-axis v) + spectral weight.
    if (tid < HH) {
        const int fu = tid % H, fv = tid / H;
        float Fre[HID] = {0.f}, Fim[HID] = {0.f};
        for (int hh = 0; hh < H; ++hh) {
            for (int ww = 0; ww < H; ++ww) {
                int ph = (fu * ww + fv * hh) & (H - 1);
                float ct = tcs[ph], st = tsn[ph];
                const float* yp = &Y[(hh * H + ww) * 9];
                #pragma unroll
                for (int c = 0; c < HID; ++c) {
                    Fre[c] += yp[c] * ct;
                    Fim[c] -= yp[c] * st;
                }
            }
        }
        int wu, wv; float sgn;
        if (fv <= H / 2) { wu = fu; wv = fv; sgn = 1.f; }
        else             { wu = (H - fu) & (H - 1); wv = H - fv; sgn = -1.f; }
        const float* wp = cw + ((long)wu * HB + wv) * (HID * 2);
        #pragma unroll
        for (int c = 0; c < HID; ++c) {
            float wr = wp[c * 2], wi = sgn * wp[c * 2 + 1];
            float fr = Fre[c] * inv, fi = Fim[c] * inv;
            Gre[tid * 9 + c] = fr * wr - fi * wi;
            Gim[tid * 9 + c] = fr * wi + fi * wr;
        }
    }
    __syncthreads();

    // Inverse DFT (real part), LN2, GELU-MLP, residual, store.
    if (tid < HH) {
        const int ww = tid % H, hh = tid / H;
        float pr[HID] = {0.f};
        for (int fv = 0; fv < H; ++fv) {
            for (int fu = 0; fu < H; ++fu) {
                int ph = (fu * ww + fv * hh) & (H - 1);
                float ct = tcs[ph], st = tsn[ph];
                const float* gr = &Gre[(fv * H + fu) * 9];
                const float* gi = &Gim[(fv * H + fu) * 9];
                #pragma unroll
                for (int c = 0; c < HID; ++c)
                    pr[c] += gr[c] * ct - gi[c] * st;
            }
        }
        float mean = 0.f;
        #pragma unroll
        for (int c = 0; c < HID; ++c) { pr[c] *= inv; mean += pr[c]; }
        mean *= 0.125f;
        float var = 0.f;
        #pragma unroll
        for (int c = 0; c < HID; ++c) { float d = pr[c] - mean; var += d * d; }
        float rstd = rsqrtf(var * 0.125f + 1e-5f);
        float t2[HID];
        #pragma unroll
        for (int c = 0; c < HID; ++c)
            t2[c] = (pr[c] - mean) * rstd * P[L2G + c] + P[L2B + c];

        float o[HID];
        #pragma unroll
        for (int j = 0; j < HID; ++j) o[j] = P[B2O + j] + raw[j];
        #pragma unroll
        for (int k = 0; k < 16; ++k) {
            float s = P[B1O + k];
            #pragma unroll
            for (int c = 0; c < HID; ++c) s += t2[c] * P[W1O + c * 16 + k];
            float g = 0.5f * s * (1.0f + erff(s * 0.70710678118f));   // exact GELU
            #pragma unroll
            for (int j = 0; j < HID; ++j) o[j] += g * P[W2O + k * HID + j];
        }
        float* dst = out + (long)(img * NTOK + tokOff + tid) * HID;
        #pragma unroll
        for (int j = 0; j < HID; ++j) dst[j] = o[j];
    }
}

// ===========================================================================
// Kernel 3: out = g @ Wu + bu  (81920x8 @ 8x768).  One WMMA per 16x16 tile
// (K=8 zero-padded into the 32-wide fragment).  B fragment hoisted & reused
// across 8 M tiles per wave; output write is the 252MB stream that bounds us.
// ===========================================================================
__global__ __launch_bounds__(128) void up_proj_kernel(
    const float* __restrict__ g, const float* __restrict__ Wu,
    const float* __restrict__ bu, float* __restrict__ out)
{
    const int tid  = threadIdx.x;
    const int wave = tid >> 5;
    const int lane = tid & 31;
    const int half = lane >> 4;
    const int nt   = blockIdx.x % 48;
    const int mg   = blockIdx.x / 48;          // 0..159
    const int n    = nt * 16 + (lane & 15);

    Frag bfrag;
    #pragma unroll
    for (int v = 0; v < 8; ++v) {
        unsigned int pv = 0;
        if (half == 0 && v < 4) {              // K = 2v, 2v+1 (only K<8 valid)
            pv = pack_bf16(Wu[(v * 2) * DIMD + n], Wu[(v * 2 + 1) * DIMD + n]);
        }
        bfrag.u[v] = pv;
    }
    const float bias = bu[n];

    const int mt0 = (mg * 4 + wave) * 8;       // 8 M-tiles per wave
    for (int t = 0; t < 8; ++t) {
        const int row0 = (mt0 + t) * 16;
        const float* grow = g + (long)(row0 + (lane & 15)) * HID;
        Frag afrag;
        #pragma unroll
        for (int v = 0; v < 8; ++v) {
            unsigned int pv = 0;
            if (half == 0 && v < 4)            // K = 2v, 2v+1 valid only
                pv = pack_bf16(grow[v * 2], grow[v * 2 + 1]);
            afrag.u[v] = pv;
        }
        v8f acc = {};
        acc = __builtin_amdgcn_wmma_f32_16x16x32_bf16(
            false, afrag.bf, false, bfrag.bf, (short)0, acc, false, false);
        #pragma unroll
        for (int i = 0; i < 8; ++i) {
            int mm = half * 8 + i;
            out[(long)(row0 + mm) * DIMD + n] = acc[i] + bias;
        }
    }
}

// ===========================================================================
extern "C" void kernel_launch(void* const* d_in, const int* in_sizes, int n_in,
                              void* d_out, int out_size, void* d_ws, size_t ws_size,
                              hipStream_t stream) {
    const float* x     = (const float*)d_in[0];
    const float* Wd    = (const float*)d_in[1];
    const float* bd    = (const float*)d_in[2];
    const float* cw_x  = (const float*)d_in[3];
    const float* cw_z  = (const float*)d_in[4];
    const float* ln1_g = (const float*)d_in[5];
    const float* ln1_b = (const float*)d_in[6];
    const float* ln2_g = (const float*)d_in[7];
    const float* ln2_b = (const float*)d_in[8];
    const float* W1    = (const float*)d_in[9];
    const float* b1    = (const float*)d_in[10];
    const float* W2    = (const float*)d_in[11];
    const float* b2    = (const float*)d_in[12];
    const float* Wu    = (const float*)d_in[13];
    const float* bu    = (const float*)d_in[14];

    float* ws1 = (float*)d_ws;                 // h      : 81920 x 8
    float* ws2 = ws1 + (size_t)MROWS * HID;    // branch : 81920 x 8

    down_proj_kernel<<<MROWS / 64, 128, 0, stream>>>(x, Wd, bd, ws1);
    branch_kernel<<<512, 256, 0, stream>>>(ws1, cw_x, cw_z, ln1_g, ln1_b,
                                           ln2_g, ln2_b, W1, b1, W2, b2, ws2);
    up_proj_kernel<<<48 * 160, 128, 0, stream>>>(ws2, Wu, bu, (float*)d_out);
}
